// Str2Str_59519656787973
// MI455X (gfx1250) — matrix operand order repaired and checked by reading, not compiled, not run on hardware
//
#include <hip/hip_runtime.h>
#include <hip/hip_bf16.h>

typedef __attribute__((ext_vector_type(16))) _Float16 v16h;
typedef __attribute__((ext_vector_type(8)))  float    v8f;

#define LSEQ   1024
#define TK     128
#define DMSA   256
#define DPAIR  128
#define DSTATE 16
#define DNODE  32
#define DEDGE  32
#define DH     64
#define EPS    1e-5f

// ---- WMMA fragment index helpers (contiguous-k 16-bit A layout used here, f32 C/D layout) ----
// A (16x32 f16): lane t<16 holds row m=t, K=h; lane t>=16: K=16+h
static __device__ __forceinline__ void a_slot(int m, int ch /*0..31*/, int &t, int &h) {
  if (ch < 16)      { t = m;      h = ch;      }
  else              { t = m + 16; h = ch - 16; }
}

static __device__ __forceinline__ v8f wmma_f16(v16h a, v16h b, v8f c) {
  return __builtin_amdgcn_wmma_f32_16x16x32_f16(false, a, false, b, (short)0, c, false, false);
}

static __device__ __forceinline__ float wave_sum(float x) {
  #pragma unroll
  for (int m = 16; m >= 1; m >>= 1) x += __shfl_xor(x, m, 32);
  return x;
}

// ---- pack B matrices into fragment order: [chunk][ntile][lane][half] ----
// B (32x16 f16): lane t<16 holds col n=t, K=h; lane t>=16: col n=t-16, K=16+h
__global__ void k_pack(const float* __restrict__ We1, const float* __restrict__ We2,
                       const float* __restrict__ Wm1,
                       _Float16* __restrict__ we1p, _Float16* __restrict__ we2p,
                       _Float16* __restrict__ wm1p) {
  int tid = threadIdx.x + blockIdx.x * blockDim.x;
  int stride = blockDim.x * gridDim.x;
  for (int e = tid; e < 4*2*32*16; e += stride) {           // We1: 128x32
    int h = e & 15, t = (e >> 4) & 31, nt = (e >> 9) & 1, ck = e >> 10;
    int n = nt*16 + (t & 15);
    int k = ck*32 + ((t < 16) ? 0 : 16) + h;
    we1p[e] = (_Float16)We1[k*32 + n];
  }
  for (int e = tid; e < 3*2*32*16; e += stride) {           // We2: 69x32 zero-padded to 96x32
    int h = e & 15, t = (e >> 4) & 31, nt = (e >> 9) & 1, ck = e >> 10;
    int n = nt*16 + (t & 15);
    int k = ck*32 + ((t < 16) ? 0 : 16) + h;
    we2p[e] = (k < 69) ? (_Float16)We2[k*32 + n] : (_Float16)0.f;
  }
  for (int e = tid; e < 3*4*32*16; e += stride) {           // Wm1: 96x64
    int h = e & 15, t = (e >> 4) & 31, nt = (e >> 9) & 3, ck = e >> 11;
    int n = nt*16 + (t & 15);
    int k = ck*32 + ((t < 16) ? 0 : 16) + h;
    wm1p[e] = (_Float16)Wm1[k*64 + n];
  }
}

// ---- node = LN( [LN(msa0), LN(state)] @ Wx + bx ); also ca, v ----
__global__ __launch_bounds__(32) void k_node(
    const float* __restrict__ msa, const float* __restrict__ state, const float* __restrict__ xyz,
    const float* __restrict__ lmw, const float* __restrict__ lmb,
    const float* __restrict__ lsw, const float* __restrict__ lsb,
    const float* __restrict__ lnw, const float* __restrict__ lnb,
    const float* __restrict__ Wx, const float* __restrict__ bx,
    float* __restrict__ node, float* __restrict__ ca, float* __restrict__ v) {
  int i = blockIdx.x, t = threadIdx.x;
  __shared__ float xin[DMSA + DSTATE];
  const float* mrow = msa + (size_t)i * DMSA;   // msa[0,0,i,:]
  float s1 = 0.f, s2 = 0.f;
  for (int c = t; c < DMSA; c += 32) { float x = mrow[c]; xin[c] = x; s1 += x; s2 += x * x; }
  s1 = wave_sum(s1); s2 = wave_sum(s2);
  float mu = s1 / DMSA, rstd = rsqrtf(s2 / DMSA - mu * mu + EPS);
  for (int c = t; c < DMSA; c += 32) xin[c] = (xin[c] - mu) * rstd * lmw[c] + lmb[c];
  float sv = (t < DSTATE) ? state[(size_t)i * DSTATE + t] : 0.f;
  float t1 = wave_sum(sv), t2 = wave_sum(sv * sv);
  float mus = t1 / DSTATE, rstds = rsqrtf(t2 / DSTATE - mus * mus + EPS);
  if (t < DSTATE) xin[DMSA + t] = (sv - mus) * rstds * lsw[t] + lsb[t];
  __syncthreads();
  float acc = bx[t];
  for (int k = 0; k < DMSA + DSTATE; ++k) acc += xin[k] * Wx[k * DNODE + t];
  float a1 = wave_sum(acc), a2 = wave_sum(acc * acc);
  float mun = a1 / DNODE, rstdn = rsqrtf(a2 / DNODE - mun * mun + EPS);
  node[(size_t)i * DNODE + t] = (acc - mun) * rstdn * lnw[t] + lnb[t];
  if (t < 3) ca[i * 3 + t] = xyz[(size_t)i * 9 + 3 + t];
  if (t < 9) { int a = t / 3, d = t % 3; v[(size_t)i * 9 + t] = xyz[(size_t)i * 9 + a * 3 + d] - xyz[(size_t)i * 9 + 3 + d]; }
}

// ---- deterministic top-k(128) per row via 9-bit radix bin select + scan ----
__global__ __launch_bounds__(1024) void k_topk(const float* __restrict__ ca,
                                               const int* __restrict__ idx,
                                               int* __restrict__ Eidx) {
  int i = blockIdx.x, j = threadIdx.x;
  __shared__ float key[LSEQ];
  __shared__ unsigned hist[512];
  __shared__ int scanbuf[LSEQ];
  __shared__ int selB, selC0;
  float dx = ca[i*3+0]-ca[j*3+0], dy = ca[i*3+1]-ca[j*3+1], dz = ca[i*3+2]-ca[j*3+2];
  float D = sqrtf(dx*dx + dy*dy + dz*dz + 1e-12f);
  int sep = idx[j] - idx[i];
  float Dg = D + ((i == j) ? 999.9f : 0.f) + 1e-6f * fabsf((float)sep);
  key[j] = Dg;
  if (j < 512) hist[j] = 0u;
  __syncthreads();
  unsigned bin = __float_as_uint(Dg) >> 22;   // positive floats: bits[30:22]
  atomicAdd(&hist[bin], 1u);
  __syncthreads();
  if (j == 0) {
    unsigned cum = 0; int B = 511;
    for (int b = 0; b < 512; ++b) { unsigned c = hist[b]; if (cum + c >= TK) { B = b; break; } cum += c; }
    selB = B; selC0 = (int)cum;
  }
  __syncthreads();
  int B = selB, c0 = selC0;
  int less = (bin < (unsigned)B) ? 1 : 0;
  scanbuf[j] = less;
  for (int off = 1; off < LSEQ; off <<= 1) {
    __syncthreads();
    int vprev = (j >= off) ? scanbuf[j - off] : 0;
    __syncthreads();
    scanbuf[j] += vprev;
  }
  __syncthreads();
  int slot = -1;
  if (less) slot = scanbuf[j] - 1;
  else if (bin == (unsigned)B) {
    int r = 0; float mine = Dg;
    for (int m = 0; m < LSEQ; ++m) {
      float o = key[m];
      if ((__float_as_uint(o) >> 22) == (unsigned)B && (o < mine || (o == mine && m < j))) ++r;
    }
    if (c0 + r < TK) slot = c0 + r;
  }
  if (slot >= 0) Eidx[i * TK + slot] = j;
}

// ---- fused edge pipeline on selected rows only: one wave = 16 rows ----
__global__ __launch_bounds__(32) void k_edge(
    const float* __restrict__ pair, const int* __restrict__ Eidx,
    const float* __restrict__ ca, const int* __restrict__ idx,
    const float* __restrict__ lpw, const float* __restrict__ lpb,
    const float* __restrict__ be1, const float* __restrict__ l1w, const float* __restrict__ l1b,
    const float* __restrict__ be2, const float* __restrict__ l2w, const float* __restrict__ l2b,
    const _Float16* __restrict__ we1p, const _Float16* __restrict__ we2p,
    float* __restrict__ efeat) {
  int t = threadIdx.x;
  int m = t & 15;
  int half = t >> 4;
  int r0 = blockIdx.x * 16;
  int row = r0 + m;
  int i = row >> 7;
  int k = row & (TK - 1);
  int j = Eidx[i * TK + k];

  __shared__ alignas(32) _Float16 aF[4][32][16];
  __shared__ alignas(32) _Float16 aF2[3][32][16];
  __shared__ float dT[16][32];

  // stage 1: LN(pair row) -> f16 A fragments (t and t+16 share row m)
  const float* src = pair + ((size_t)i * LSEQ + j) * DPAIR + half * 64;
  float buf[64];
  float s1 = 0.f, s2 = 0.f;
  #pragma unroll
  for (int c = 0; c < 64; ++c) { float x = src[c]; buf[c] = x; s1 += x; s2 += x * x; }
  s1 += __shfl_xor(s1, 16, 32);
  s2 += __shfl_xor(s2, 16, 32);
  float mu = s1 / DPAIR, rstd = rsqrtf(s2 / DPAIR - mu * mu + EPS);
  #pragma unroll
  for (int c = 0; c < 64; ++c) {
    int ch = half * 64 + c;
    float val = (buf[c] - mu) * rstd * lpw[ch] + lpb[ch];
    int tt, hh; a_slot(m, ch & 31, tt, hh);
    aF[ch >> 5][tt][hh] = (_Float16)val;
  }
  __syncthreads();

  // GEMM1: (16x128) @ We1(128x32), K chunks of 32, two N tiles
  v8f acc0 = {}; v8f acc1 = {};
  #pragma unroll
  for (int ck = 0; ck < 4; ++ck) {
    v16h a  = *(const v16h*)&aF[ck][t][0];
    v16h b0 = *(const v16h*)&we1p[((ck * 2 + 0) * 32 + t) * 16];
    v16h b1 = *(const v16h*)&we1p[((ck * 2 + 1) * 32 + t) * 16];
    acc0 = wmma_f16(a, b0, acc0);
    acc1 = wmma_f16(a, b1, acc1);
  }
  {
    const float* a0 = (const float*)&acc0;
    const float* a1 = (const float*)&acc1;
    int n = t & 15, rb = (t < 16) ? 0 : 8;
    #pragma unroll
    for (int r = 0; r < 8; ++r) {
      dT[rb + r][n]      = a0[r] + be1[n];
      dT[rb + r][16 + n] = a1[r] + be1[16 + n];
    }
  }
  __syncthreads();

  // stage 2 inputs: [LN_e1(p) (32) | rbf(D) (36) | neigh (1) | zeros -> 96]
  if (t < 16) {
    float u1 = 0.f, u2 = 0.f;
    #pragma unroll
    for (int c = 0; c < 32; ++c) { float x = dT[t][c]; u1 += x; u2 += x * x; }
    float muv = u1 / 32.f, rs = rsqrtf(u2 / 32.f - muv * muv + EPS);
    #pragma unroll
    for (int c = 0; c < 32; ++c) {
      float val = (dT[t][c] - muv) * rs * l1w[c] + l1b[c];
      int tt, hh; a_slot(t, c, tt, hh);
      aF2[0][tt][hh] = (_Float16)val;
    }
  } else {
    int mm = t - 16;
    int rr = r0 + mm;
    int ii = rr >> 7, kk2 = rr & (TK - 1);
    int jj = Eidx[ii * TK + kk2];
    float dx = ca[ii*3+0]-ca[jj*3+0], dy = ca[ii*3+1]-ca[jj*3+1], dz = ca[ii*3+2]-ca[jj*3+2];
    float D = sqrtf(dx*dx + dy*dy + dz*dz + 1e-12f);
    const float sig = 20.f / 36.f;
    #pragma unroll
    for (int rb = 0; rb < 36; ++rb) {
      float dmu = 2.f + 20.f * rb / 35.f;
      float z = (D - dmu) / sig;
      int ch = 32 + rb;
      int tt, hh; a_slot(mm, ch & 31, tt, hh);
      aF2[ch >> 5][tt][hh] = (_Float16)expf(-z * z);
    }
    int sep = idx[jj] - idx[ii];
    float neigh = (sep == 1) ? 1.f : ((sep == -1) ? -1.f : 0.f);
    { int tt, hh; a_slot(mm, 68 & 31, tt, hh); aF2[2][tt][hh] = (_Float16)neigh; }
    #pragma unroll
    for (int ch = 69; ch < 96; ++ch) {
      int tt, hh; a_slot(mm, ch & 31, tt, hh);
      aF2[ch >> 5][tt][hh] = (_Float16)0.f;
    }
  }
  __syncthreads();

  // GEMM2: (16x96) @ We2pad(96x32)
  v8f e0 = {}; v8f e1v = {};
  #pragma unroll
  for (int ck = 0; ck < 3; ++ck) {
    v16h a  = *(const v16h*)&aF2[ck][t][0];
    v16h b0 = *(const v16h*)&we2p[((ck * 2 + 0) * 32 + t) * 16];
    v16h b1 = *(const v16h*)&we2p[((ck * 2 + 1) * 32 + t) * 16];
    e0  = wmma_f16(a, b0, e0);
    e1v = wmma_f16(a, b1, e1v);
  }
  {
    const float* a0 = (const float*)&e0;
    const float* a1 = (const float*)&e1v;
    int n = t & 15, rb = (t < 16) ? 0 : 8;
    #pragma unroll
    for (int r = 0; r < 8; ++r) {
      dT[rb + r][n]      = a0[r] + be2[n];
      dT[rb + r][16 + n] = a1[r] + be2[16 + n];
    }
  }
  __syncthreads();
  if (t < 16) {
    float u1 = 0.f, u2 = 0.f;
    #pragma unroll
    for (int c = 0; c < 32; ++c) { float x = dT[t][c]; u1 += x; u2 += x * x; }
    float muv = u1 / 32.f, rs = rsqrtf(u2 / 32.f - muv * muv + EPS);
    float* dst = efeat + (size_t)(r0 + t) * DEDGE;
    #pragma unroll
    for (int c = 0; c < 32; ++c) dst[c] = (dT[t][c] - muv) * rs * l2w[c] + l2b[c];
  }
}

// ---- message MLP (WMMA 96->64, relu) + k-reductions + Rodrigues update ----
__global__ __launch_bounds__(128) void k_msg(
    const int* __restrict__ Eidx, const float* __restrict__ node, const float* __restrict__ efeat,
    const float* __restrict__ ca, const float* __restrict__ v,
    const _Float16* __restrict__ wm1p,
    const float* __restrict__ bm1, const float* __restrict__ Wm2, const float* __restrict__ bm2,
    const float* __restrict__ Wm3, const float* __restrict__ bm3,
    float* __restrict__ out) {
  int i = blockIdx.x;
  int tid = threadIdx.x;

  __shared__ alignas(32) _Float16 aF[8][3][32][16];
  __shared__ float hid[TK][DH];
  __shared__ float red[TK][6];
  __shared__ float hbar[DH];
  __shared__ float offv[6];
  __shared__ float w3[DH * 8];
  __shared__ float w2[DH * 16];

  for (int e = tid; e < DH * 8;  e += 128) w3[e] = Wm3[e];
  for (int e = tid; e < DH * 16; e += 128) w2[e] = Wm2[e];

  int k = tid;
  int j = Eidx[i * TK + k];
  int mt = k >> 4, m = k & 15;
  const float* hi = node + (size_t)i * DNODE;
  const float* hj = node + (size_t)j * DNODE;
  const float* ef = efeat + ((size_t)i * TK + k) * DEDGE;
  for (int c = 0; c < 96; ++c) {
    float val = (c < 32) ? hi[c] : ((c < 64) ? hj[c - 32] : ef[c - 64]);
    int tt, hh; a_slot(m, c & 31, tt, hh);
    aF[mt][c >> 5][tt][hh] = (_Float16)val;
  }
  __syncthreads();

  int w = tid >> 5, t = tid & 31;
  const v8f vzero = {};
  for (int q = 0; q < 2; ++q) {
    int mtile = w * 2 + q;
    v8f acc[4] = {vzero, vzero, vzero, vzero};
    for (int ck = 0; ck < 3; ++ck) {
      v16h a = *(const v16h*)&aF[mtile][ck][t][0];
      #pragma unroll
      for (int nt = 0; nt < 4; ++nt) {
        v16h b = *(const v16h*)&wm1p[((ck * 4 + nt) * 32 + t) * 16];
        acc[nt] = wmma_f16(a, b, acc[nt]);
      }
    }
    int n = t & 15, rbase = (t < 16) ? 0 : 8;
    #pragma unroll
    for (int nt = 0; nt < 4; ++nt) {
      const float* ap = (const float*)&acc[nt];
      int c = nt * 16 + n;
      float bias = bm1[c];
      #pragma unroll
      for (int r = 0; r < 8; ++r) {
        float val = ap[r] + bias;
        hid[mtile * 16 + rbase + r][c] = (val > 0.f) ? val : 0.f;
      }
    }
  }
  __syncthreads();

  // per-k: g = hid@Wm3 + bm3, contribution to offset via rhat and v_j
  {
    float g[8];
    #pragma unroll
    for (int o = 0; o < 8; ++o) g[o] = bm3[o];
    for (int h = 0; h < DH; ++h) {
      float hv = hid[k][h];
      #pragma unroll
      for (int o = 0; o < 8; ++o) g[o] += hv * w3[h * 8 + o];
    }
    float rel0 = ca[j*3+0]-ca[i*3+0], rel1 = ca[j*3+1]-ca[i*3+1], rel2 = ca[j*3+2]-ca[i*3+2];
    float rn = sqrtf(rel0*rel0 + rel1*rel1 + rel2*rel2);
    float inv = 1.f / (rn + 1e-8f);
    float rhat[3] = {rel0 * inv, rel1 * inv, rel2 * inv};
    const float* vj = v + (size_t)j * 9;
    #pragma unroll
    for (int c = 0; c < 2; ++c)
      #pragma unroll
      for (int d = 0; d < 3; ++d) {
        float s = g[c * 4] * rhat[d];
        #pragma unroll
        for (int mm = 0; mm < 3; ++mm) s += g[c * 4 + 1 + mm] * vj[mm * 3 + d];
        red[k][c * 3 + d] = s;
      }
  }
  __syncthreads();
  if (tid < DH) {
    float s = 0.f;
    for (int kk = 0; kk < TK; ++kk) s += hid[kk][tid];
    hbar[tid] = s / TK;
  }
  __syncthreads();
  if (tid < 6) {
    float s = 0.f;
    for (int kk = 0; kk < TK; ++kk) s += red[kk][tid];
    offv[tid] = s / TK;
  }
  __syncthreads();
  if (tid < 16) {
    float s = bm2[tid];
    for (int h = 0; h < DH; ++h) s += hbar[h] * w2[h * 16 + tid];
    out[LSEQ * 9 + i * 16 + tid] = s;
  } else if (tid < 19) {
    int a = tid - 16;
    float T0 = offv[0] / 10.f,  T1 = offv[1] / 10.f,  T2 = offv[2] / 10.f;
    float R0 = offv[3] / 100.f, R1 = offv[4] / 100.f, R2 = offv[5] / 100.f;
    float Rang = sqrtf(R0*R0 + R1*R1 + R2*R2);
    float rinv = 1.f / (Rang + EPS);
    float Rv0 = R0 * rinv, Rv1 = R1 * rinv, Rv2 = R2 * rinv;
    const float* va = v + (size_t)i * 9 + a * 3;
    float Rdv = Rv0*va[0] + Rv1*va[1] + Rv2*va[2];
    float cx = Rv1*va[2] - Rv2*va[1];
    float cy = Rv2*va[0] - Rv0*va[2];
    float cz = Rv0*va[1] - Rv1*va[0];
    float cA = cosf(Rang), sA = sinf(Rang);
    out[(i*3+a)*3+0] = (va[0]-Rv0*Rdv)*cA + cx*sA + Rv0*Rdv + ca[i*3+0] + T0;
    out[(i*3+a)*3+1] = (va[1]-Rv1*Rdv)*cA + cy*sA + Rv1*Rdv + ca[i*3+1] + T1;
    out[(i*3+a)*3+2] = (va[2]-Rv2*Rdv)*cA + cz*sA + Rv2*Rdv + ca[i*3+2] + T2;
  }
}

extern "C" void kernel_launch(void* const* d_in, const int* in_sizes, int n_in,
                              void* d_out, int out_size, void* d_ws, size_t ws_size,
                              hipStream_t stream) {
  (void)in_sizes; (void)n_in; (void)out_size; (void)ws_size;
  const float* msa   = (const float*)d_in[0];
  const float* pair  = (const float*)d_in[1];
  const float* xyz   = (const float*)d_in[2];
  const float* state = (const float*)d_in[3];
  const int*   idx   = (const int*)d_in[4];
  // d_in[5] = top_k (128, hardcoded)
  const float* lmw = (const float*)d_in[6];
  const float* lmb = (const float*)d_in[7];
  const float* lpw = (const float*)d_in[8];
  const float* lpb = (const float*)d_in[9];
  const float* lsw = (const float*)d_in[10];
  const float* lsb = (const float*)d_in[11];
  const float* lnw = (const float*)d_in[12];
  const float* lnb = (const float*)d_in[13];
  const float* l1w = (const float*)d_in[14];
  const float* l1b = (const float*)d_in[15];
  const float* l2w = (const float*)d_in[16];
  const float* l2b = (const float*)d_in[17];
  const float* Wx  = (const float*)d_in[18];
  const float* bx  = (const float*)d_in[19];
  const float* We1 = (const float*)d_in[20];
  const float* be1 = (const float*)d_in[21];
  const float* We2 = (const float*)d_in[22];
  const float* be2 = (const float*)d_in[23];
  const float* Wm1 = (const float*)d_in[24];
  const float* bm1 = (const float*)d_in[25];
  const float* Wm2 = (const float*)d_in[26];
  const float* bm2 = (const float*)d_in[27];
  const float* Wm3 = (const float*)d_in[28];
  const float* bm3 = (const float*)d_in[29];
  float* out = (float*)d_out;

  char* ws = (char*)d_ws;
  size_t off = 0;
  auto alloc = [&](size_t bytes) -> void* {
    void* p = ws + off;
    off += bytes;
    off = (off + 255) & ~(size_t)255;
    return p;
  };
  float*    node  = (float*)alloc((size_t)LSEQ * DNODE * 4);
  float*    ca    = (float*)alloc((size_t)LSEQ * 3 * 4);
  float*    vv    = (float*)alloc((size_t)LSEQ * 9 * 4);
  int*      Eidx  = (int*)alloc((size_t)LSEQ * TK * 4);
  float*    efeat = (float*)alloc((size_t)LSEQ * TK * DEDGE * 4);
  _Float16* we1p  = (_Float16*)alloc(4 * 2 * 32 * 16 * 2);
  _Float16* we2p  = (_Float16*)alloc(3 * 2 * 32 * 16 * 2);
  _Float16* wm1p  = (_Float16*)alloc(3 * 4 * 32 * 16 * 2);

  k_pack<<<1, 256, 0, stream>>>(We1, We2, Wm1, we1p, we2p, wm1p);
  k_node<<<LSEQ, 32, 0, stream>>>(msa, state, xyz, lmw, lmb, lsw, lsb, lnw, lnb, Wx, bx, node, ca, vv);
  k_topk<<<LSEQ, LSEQ, 0, stream>>>(ca, idx, Eidx);
  k_edge<<<(LSEQ * TK) / 16, 32, 0, stream>>>(pair, Eidx, ca, idx, lpw, lpb,
                                              be1, l1w, l1b, be2, l2w, l2b, we1p, we2p, efeat);
  k_msg<<<LSEQ, 128, 0, stream>>>(Eidx, node, efeat, ca, vv, wm1p, bm1, Wm2, bm2, Wm3, bm3, out);
}
